// SpatialAttentionLayer_37830071943782
// MI455X (gfx1250) — matrix operand order repaired and checked by reading, not compiled, not run on hardware
//
#include <hip/hip_runtime.h>
#include <hip/hip_bf16.h>
#include <math.h>

typedef __attribute__((ext_vector_type(16))) _Float16 v16h;
typedef __attribute__((ext_vector_type(8)))  _Float16 v8h;
typedef __attribute__((ext_vector_type(8)))  float    v8f;

#define B_DIM 64
#define N_DIM 512
#define F_DIM 16
#define T_DIM 288

// ---------------------------------------------------------------------------
// WMMA fragment loaders (row-major f16 memory, NT gemm: C[r,c] = sum_k A[r,k]*B[c,k])
// A-matrix 16x32 f16 k order used here (contiguous, same as B):
//   lane<16 : m=lane,    elems 0-15 = K k0..k0+15
//   lane>=16: m=lane-16, elems 0-15 = K k0+16..k0+31
// -> p = row + k0 + 16*(lane>=16); lo = p[0..7], hi = p[8..15]
__device__ __forceinline__ v16h load_frag_a(const _Float16* p) {
    v8h lo = *(const v8h*)(p);
    v8h hi = *(const v8h*)(p + 8);
    return __builtin_shufflevector(lo, hi, 0,1,2,3,4,5,6,7,8,9,10,11,12,13,14,15);
}
// B-matrix 32x16 f16 k order used here (same as A): lane-half n=lane%16 holds
// 16 contiguous K values: lane<16 -> K k0..k0+15, lane>=16 -> K k0+16..k0+31.
// Our B column c is row c of the second (row-major-in-k) operand.
__device__ __forceinline__ v16h load_frag_b(const _Float16* p) {
    v8h lo = *(const v8h*)(p);
    v8h hi = *(const v8h*)(p + 8);
    return __builtin_shufflevector(lo, hi, 0,1,2,3,4,5,6,7,8,9,10,11,12,13,14,15);
}

__device__ __forceinline__ v8f wmma_f16(v16h a, v16h b, v8f c) {
    return __builtin_amdgcn_wmma_f32_16x16x32_f16(false, a, false, b, (short)0, c, false, false);
}

// ---------------------------------------------------------------------------
// Kernel 1: single streaming pass over x.
//   rhsT[b,n,t] = sum_f x[b,n,f,t] * W3[f]        (f16)
//   xw1[f]     = sum_t x[b,n,f,t] * W1[t]
//   lhs[b,n,t] = sum_f xw1[f] * W2[f,t]           (f16)
// one block per (n,b), 288 threads (9 waves)
__global__ __launch_bounds__(288)
void k_reduce(const float* __restrict__ x, const float* __restrict__ W1,
              const float* __restrict__ W2, const float* __restrict__ W3,
              _Float16* __restrict__ rhsT, _Float16* __restrict__ lhsg) {
    __shared__ float tile[T_DIM * 17];   // [t][f], padded stride 17
    __shared__ float w1s[T_DIM];
    __shared__ float w3s[F_DIM];
    __shared__ float part[16 * 16];
    __shared__ float xw1[F_DIM];

    const int n = blockIdx.x;
    const int b = blockIdx.y;
    const int t = threadIdx.x;

    const float* xb = x + (((size_t)b * N_DIM + n) * F_DIM) * T_DIM;
    w1s[t] = W1[t];
    if (t < F_DIM) w3s[t] = W3[t];
#pragma unroll
    for (int f = 0; f < F_DIM; ++f)
        tile[t * 17 + f] = xb[(size_t)f * T_DIM + t];
    __syncthreads();

    // rhsT (reduce over f)
    float r = 0.f;
#pragma unroll
    for (int f = 0; f < F_DIM; ++f) r += tile[t * 17 + f] * w3s[f];
    rhsT[((size_t)b * N_DIM + n) * T_DIM + t] = (_Float16)r;

    // xw1 partials: thread j<256 -> f = j&15, chunk = j>>4 (18 t's each)
    if (t < 256) {
        const int f = t & 15, c = t >> 4;
        float p = 0.f;
        for (int tt = c * 18; tt < c * 18 + 18; ++tt)
            p += tile[tt * 17 + f] * w1s[tt];
        part[f * 16 + c] = p;
    }
    __syncthreads();
    if (t < F_DIM) {
        float s = 0.f;
        for (int c = 0; c < 16; ++c) s += part[t * 16 + c];
        xw1[t] = s;
    }
    __syncthreads();

    // lhs = xw1 @ W2  (tiny K=16 gemv per t)
    float l = 0.f;
#pragma unroll
    for (int f = 0; f < F_DIM; ++f) l += xw1[f] * W2[f * T_DIM + t];
    lhsg[((size_t)b * N_DIM + n) * T_DIM + t] = (_Float16)l;
}

// ---------------------------------------------------------------------------
// Kernel 2: product = lhs(512x288) x rhsT^T, +b_s, sigmoid -> sig (f16)
// block = 256 threads = 8 waves (4 row-waves x 2 col-waves), block tile 128x64
__global__ __launch_bounds__(256)
void k_gemm_sig(const _Float16* __restrict__ lhs, const _Float16* __restrict__ rhsT,
                const float* __restrict__ b_s, _Float16* __restrict__ sig) {
    const int b    = blockIdx.z;
    const int tid  = threadIdx.x;
    const int wave = tid >> 5, lane = tid & 31;
    const int wm = wave & 3, wn = wave >> 2;
    const int rowBase = blockIdx.y * 128 + wm * 32;
    const int colBase = blockIdx.x * 64  + wn * 32;
    const int laneLo = lane & 15, hi16 = lane >> 4;

    const _Float16* Ab = lhs  + (size_t)b * N_DIM * T_DIM;
    const _Float16* Bb = rhsT + (size_t)b * N_DIM * T_DIM;
    const _Float16* arow0 = Ab + (size_t)(rowBase + laneLo) * T_DIM + hi16 * 16;
    const _Float16* arow1 = arow0 + (size_t)16 * T_DIM;
    const _Float16* bcol0 = Bb + (size_t)(colBase + laneLo) * T_DIM + hi16 * 16;
    const _Float16* bcol1 = bcol0 + (size_t)16 * T_DIM;

    v8f acc00 = {}, acc01 = {}, acc10 = {}, acc11 = {};
#pragma unroll
    for (int k0 = 0; k0 < T_DIM; k0 += 32) {
        v16h a0 = load_frag_a(arow0 + k0);
        v16h a1 = load_frag_a(arow1 + k0);
        v16h b0 = load_frag_b(bcol0 + k0);
        v16h b1 = load_frag_b(bcol1 + k0);
        acc00 = wmma_f16(a0, b0, acc00);
        acc01 = wmma_f16(a0, b1, acc01);
        acc10 = wmma_f16(a1, b0, acc10);
        acc11 = wmma_f16(a1, b1, acc11);
    }

    const int c0 = colBase + laneLo, c1 = c0 + 16;
#pragma unroll
    for (int v = 0; v < 8; ++v) {
        const int r0 = rowBase + v + hi16 * 8;
        const int r1 = r0 + 16;
        float x00 = acc00[v] + b_s[(size_t)r0 * N_DIM + c0];
        float x01 = acc01[v] + b_s[(size_t)r0 * N_DIM + c1];
        float x10 = acc10[v] + b_s[(size_t)r1 * N_DIM + c0];
        float x11 = acc11[v] + b_s[(size_t)r1 * N_DIM + c1];
        sig[((size_t)b * N_DIM + r0) * N_DIM + c0] = (_Float16)(1.f / (1.f + __expf(-x00)));
        sig[((size_t)b * N_DIM + r0) * N_DIM + c1] = (_Float16)(1.f / (1.f + __expf(-x01)));
        sig[((size_t)b * N_DIM + r1) * N_DIM + c0] = (_Float16)(1.f / (1.f + __expf(-x10)));
        sig[((size_t)b * N_DIM + r1) * N_DIM + c1] = (_Float16)(1.f / (1.f + __expf(-x11)));
    }
}

// ---------------------------------------------------------------------------
// Kernel 3: s[b,n,i] = sum_j sig[b,n,j] * V_s[i,j]  (K=512, f32 out to d_out)
__global__ __launch_bounds__(256)
void k_gemm_out(const _Float16* __restrict__ sig, const _Float16* __restrict__ Vh,
                float* __restrict__ out) {
    const int b    = blockIdx.z;
    const int tid  = threadIdx.x;
    const int wave = tid >> 5, lane = tid & 31;
    const int wm = wave & 3, wn = wave >> 2;
    const int rowBase = blockIdx.y * 128 + wm * 32;
    const int colBase = blockIdx.x * 64  + wn * 32;
    const int laneLo = lane & 15, hi16 = lane >> 4;

    const _Float16* Ab = sig + (size_t)b * N_DIM * N_DIM;
    const _Float16* arow0 = Ab + (size_t)(rowBase + laneLo) * N_DIM + hi16 * 16;
    const _Float16* arow1 = arow0 + (size_t)16 * N_DIM;
    const _Float16* bcol0 = Vh + (size_t)(colBase + laneLo) * N_DIM + hi16 * 16;
    const _Float16* bcol1 = bcol0 + (size_t)16 * N_DIM;

    v8f acc00 = {}, acc01 = {}, acc10 = {}, acc11 = {};
    for (int k0 = 0; k0 < N_DIM; k0 += 32) {
        v16h a0 = load_frag_a(arow0 + k0);
        v16h a1 = load_frag_a(arow1 + k0);
        v16h b0 = load_frag_b(bcol0 + k0);
        v16h b1 = load_frag_b(bcol1 + k0);
        acc00 = wmma_f16(a0, b0, acc00);
        acc01 = wmma_f16(a0, b1, acc01);
        acc10 = wmma_f16(a1, b0, acc10);
        acc11 = wmma_f16(a1, b1, acc11);
    }

    const int c0 = colBase + laneLo, c1 = c0 + 16;
#pragma unroll
    for (int v = 0; v < 8; ++v) {
        const int r0 = rowBase + v + hi16 * 8;
        const int r1 = r0 + 16;
        out[((size_t)b * N_DIM + r0) * N_DIM + c0] = acc00[v];
        out[((size_t)b * N_DIM + r0) * N_DIM + c1] = acc01[v];
        out[((size_t)b * N_DIM + r1) * N_DIM + c0] = acc10[v];
        out[((size_t)b * N_DIM + r1) * N_DIM + c1] = acc11[v];
    }
}

// ---------------------------------------------------------------------------
// Kernel 4: V_s f32 -> f16
__global__ __launch_bounds__(256)
void k_cvtV(const float* __restrict__ V, _Float16* __restrict__ Vh) {
    const int i = blockIdx.x * 256 + threadIdx.x;
    Vh[i] = (_Float16)V[i];
}

// ---------------------------------------------------------------------------
// Kernel 5: softmax over axis 1 (over n, per (b,i)), in-place in d_out.
// thread owns one column i of batch b; reads coalesced across lanes.
__global__ __launch_bounds__(256)
void k_softmax(float* __restrict__ s) {
    const int b = blockIdx.y;
    const int i = blockIdx.x * 256 + threadIdx.x;
    float* base = s + (size_t)b * N_DIM * N_DIM + i;
    float m = -INFINITY;
    for (int n = 0; n < N_DIM; ++n) m = fmaxf(m, base[(size_t)n * N_DIM]);
    float sum = 0.f;
    for (int n = 0; n < N_DIM; ++n) sum += __expf(base[(size_t)n * N_DIM] - m);
    const float inv = 1.f / sum;
    for (int n = 0; n < N_DIM; ++n) {
        float v = base[(size_t)n * N_DIM];
        base[(size_t)n * N_DIM] = __expf(v - m) * inv;
    }
}

// ---------------------------------------------------------------------------
extern "C" void kernel_launch(void* const* d_in, const int* in_sizes, int n_in,
                              void* d_out, int out_size, void* d_ws, size_t ws_size,
                              hipStream_t stream) {
    const float* x   = (const float*)d_in[0];
    const float* W1  = (const float*)d_in[1];
    const float* W2  = (const float*)d_in[2];
    const float* W3  = (const float*)d_in[3];
    const float* b_s = (const float*)d_in[4];
    const float* V_s = (const float*)d_in[5];
    float* out = (float*)d_out;

    char* ws = (char*)d_ws;
    const size_t szRhs = (size_t)B_DIM * N_DIM * T_DIM * sizeof(_Float16); // 18.9 MB
    const size_t szSig = (size_t)B_DIM * N_DIM * N_DIM * sizeof(_Float16); // 33.6 MB
    _Float16* rhsT = (_Float16*)(ws);
    _Float16* lhsg = (_Float16*)(ws + szRhs);
    _Float16* sig  = (_Float16*)(ws + 2 * szRhs);
    _Float16* Vh   = (_Float16*)(ws + 2 * szRhs + szSig);

    // V_s -> f16 (512*512 elements)
    k_cvtV<<<(N_DIM * N_DIM) / 256, 256, 0, stream>>>(V_s, Vh);

    // streaming pass over x: rhsT + lhs
    k_reduce<<<dim3(N_DIM, B_DIM), T_DIM, 0, stream>>>(x, W1, W2, W3, rhsT, lhsg);

    // product -> sigmoid (f16)
    k_gemm_sig<<<dim3(N_DIM / 64, N_DIM / 128, B_DIM), 256, 0, stream>>>(lhsg, rhsT, b_s, sig);

    // sigmoid x V_s^T -> s (f32, into d_out)
    k_gemm_out<<<dim3(N_DIM / 64, N_DIM / 128, B_DIM), 256, 0, stream>>>(sig, Vh, out);

    // softmax over axis 1, in place
    k_softmax<<<dim3(N_DIM / 256, B_DIM), 256, 0, stream>>>(out);
}
